// Decoder_37692632989988
// MI455X (gfx1250) — matrix operand order repaired and checked by reading, not compiled, not run on hardware
//
#include <hip/hip_runtime.h>

// ---------------------------------------------------------------- constants
#define LNUM 3
#define HDIM 64
#define NNODES 100000
#define BPAIR 10000
#define KNEG 16
#define D_IN 576          // L*L*H
#define D_H1 192          // L*H
#define NROWS (BPAIR * (KNEG + 1))   // 170000
#define NUNITS ((NROWS + 31) / 32)   // 5313 (last unit half-real, stores guarded)
#define WAVES_PER_BLOCK 2

// LDS strides (bank-conflict avoidance, 64 banks x 4B)
#define S_STRIDE  200     // bf16 units/row: 400B = 100 words, 16 rows distinct banks
#define H1_STRIDE 200     // bf16 units/row
#define H2_STRIDE 69      // f32/row: 69 % 64 == 5 (odd) -> 32 rows distinct banks

typedef __attribute__((ext_vector_type(16))) __bf16 v16bf;
typedef __attribute__((ext_vector_type(4)))  __bf16 v4bf;
typedef __attribute__((ext_vector_type(8)))  float  v8f;

__device__ __forceinline__ __bf16 f2bf(float f) { return (__bf16)f; }

// ------------------------------------------------------------- weight prep
// Swizzle W1 (192x576 row-major f32) into bf16 B-fragment-major layout.
// Fragment (kt,nt): 32 lanes x 16 bf16; element e of lane l is
// W1[n, k], n = nt*16 + (l&15), k = kt*32 + (l>=16 ? 16 : 0) + e
// (contiguous k order; the A fragments use the same order).
__global__ void build_w1frag(const float* __restrict__ W1, __bf16* __restrict__ out) {
    int i = blockIdx.x * blockDim.x + threadIdx.x;
    if (i >= 18 * 12 * 512) return;
    int e    = i & 15;
    int lane = (i >> 4) & 31;
    int frag = i >> 9;              // kt*12 + nt
    int nt = frag % 12;
    int kt = frag / 12;
    int n = nt * 16 + (lane & 15);
    int k = kt * 32 + ((lane >> 4) << 4) + e;
    out[i] = (__bf16)W1[n * D_IN + k];
}

__global__ void build_w2frag(const float* __restrict__ W2, __bf16* __restrict__ out) {
    int i = blockIdx.x * blockDim.x + threadIdx.x;
    if (i >= 6 * 4 * 512) return;
    int e    = i & 15;
    int lane = (i >> 4) & 31;
    int frag = i >> 9;              // kt*4 + nt
    int nt = frag % 4;
    int kt = frag / 4;
    int n = nt * 16 + (lane & 15);
    int k = kt * 32 + ((lane >> 4) << 4) + e;
    out[i] = (__bf16)W2[n * D_H1 + k];
}

// ----------------------------------------------------------- main MLP pass
// One wave per 32-row unit (two 16-row WMMA tiles sharing every B fragment):
// gather pair embeddings (bf16) -> on-the-fly A fragments -> WMMA layer1
// (12 N-tiles x 18 K-steps x 2 M-tiles = 432 WMMAs/unit) -> LDS -> WMMA
// layer2 (4 x 6 x 2) -> dot w3 -> exp(sigmoid(.)) scores.
__global__ __launch_bounds__(64)
void pair_mlp_kernel(const float* __restrict__ embeds,
                     const int*   __restrict__ pos,
                     const int*   __restrict__ neg,
                     const float* __restrict__ b1,
                     const float* __restrict__ b2,
                     const float* __restrict__ W3,
                     const float* __restrict__ b3,
                     const __bf16* __restrict__ w1frag,
                     const __bf16* __restrict__ w2frag,
                     float* __restrict__ scores)
{
    __shared__ __align__(16) __bf16 s_src[WAVES_PER_BLOCK][32 * S_STRIDE];
    __shared__ __align__(16) __bf16 s_dst[WAVES_PER_BLOCK][32 * S_STRIDE];
    __shared__ __align__(16) __bf16 s_h1 [WAVES_PER_BLOCK][32 * H1_STRIDE];
    __shared__ __align__(16) float  s_h2 [WAVES_PER_BLOCK][32 * H2_STRIDE];
    __shared__               int    s_nod[WAVES_PER_BLOCK][64];

    const int wave = threadIdx.x >> 5;
    const int lane = threadIdx.x & 31;
    const int unit = blockIdx.x * WAVES_PER_BLOCK + wave;
    if (unit >= NUNITS) return;   // waves independent; no block barriers used

    // --- row -> pair node ids (rows [0,B) = pos, rows [B,..) = neg flat b*K+k)
    {
        int r = unit * 32 + lane;
        if (r >= NROWS) r = NROWS - 1;            // clamp: tail unit dup-computes
        const int* p = (r < BPAIR) ? (pos + 2 * r) : (neg + 2 * (r - BPAIR));
        s_nod[wave][lane]      = p[0];
        s_nod[wave][32 + lane] = p[1];
    }
    asm volatile("s_wait_dscnt 0" ::: "memory");

    // --- gather: 32 rows x (L*H=192) floats for src/dst, stored as bf16
    #pragma unroll 1
    for (int t = lane; t < 32 * 48; t += 32) {     // 48 float4 chunks per row
        int m   = t / 48;
        int rem = t - m * 48;
        int l   = rem >> 4;                         // layer 0..2
        int c   = rem & 15;                         // float4 within 64-float seg
        int na = s_nod[wave][m];
        int nb = s_nod[wave][32 + m];
        float4 va = *(const float4*)(embeds + ((size_t)l * NNODES + na) * HDIM + c * 4);
        float4 vb = *(const float4*)(embeds + ((size_t)l * NNODES + nb) * HDIM + c * 4);
        v4bf pa = { f2bf(va.x), f2bf(va.y), f2bf(va.z), f2bf(va.w) };
        v4bf pb = { f2bf(vb.x), f2bf(vb.y), f2bf(vb.z), f2bf(vb.w) };
        *(v4bf*)&s_src[wave][m * S_STRIDE + l * 64 + c * 4] = pa;
        *(v4bf*)&s_dst[wave][m * S_STRIDE + l * 64 + c * 4] = pb;
    }
    asm volatile("s_wait_dscnt 0" ::: "memory");

    const int m   = lane & 15;
    const int hi8 = (lane >> 4) << 3;               // 0/8: C-layout M shift; hi8<<1 is the A-layout K shift
    const __bf16* s0 = &s_src[wave][ m       * S_STRIDE];  // tile0 row
    const __bf16* d0 = &s_dst[wave][ m       * S_STRIDE];
    const __bf16* s1 = &s_src[wave][(m + 16) * S_STRIDE];  // tile1 row
    const __bf16* d1 = &s_dst[wave][(m + 16) * S_STRIDE];

    const v8f vzero = {0.f, 0.f, 0.f, 0.f, 0.f, 0.f, 0.f, 0.f};

    // ---------------- layer 1: [32x576] x [576x192]
    v8f acc0[12], acc1[12];
    #pragma unroll
    for (int i = 0; i < 12; ++i) { acc0[i] = vzero; acc1[i] = vzero; }

    #pragma unroll 1
    for (int kt = 0; kt < 18; ++kt) {
        v16bf a0, a1;
        #pragma unroll
        for (int e = 0; e < 16; ++e) {
            // A 16x32, matching B fragment order: k_local = e + (lane>=16 ? 16:0)
            int kl = e + (hi8 << 1);
            int k  = kt * 32 + kl;                  // feature index, < 576
            int l1 = k / 192;
            int rem = k - l1 * 192;
            int l2 = rem >> 6;
            int h  = rem & 63;
            int is = l1 * 64 + h, id = l2 * 64 + h;
            a0[e] = f2bf((float)s0[is] * (float)d0[id]);
            a1[e] = f2bf((float)s1[is] * (float)d1[id]);
        }
        #pragma unroll
        for (int nt = 0; nt < 12; ++nt) {
            v16bf b = *(const v16bf*)(w1frag + (size_t)(kt * 12 + nt) * 512 + lane * 16);
            acc0[nt] = __builtin_amdgcn_wmma_f32_16x16x32_bf16(
                false, a0, false, b, (short)0, acc0[nt], false, false);
            acc1[nt] = __builtin_amdgcn_wmma_f32_16x16x32_bf16(
                false, a1, false, b, (short)0, acc1[nt], false, false);
        }
    }

    // bias + relu -> h1 (bf16); C/D layout: VGPR r is row (r+hi8), col nt*16+m
    #pragma unroll
    for (int nt = 0; nt < 12; ++nt) {
        int n = nt * 16 + m;
        float bias = b1[n];
        #pragma unroll
        for (int r = 0; r < 8; ++r) {
            s_h1[wave][(r + hi8)      * H1_STRIDE + n] = f2bf(fmaxf(acc0[nt][r] + bias, 0.0f));
            s_h1[wave][(r + hi8 + 16) * H1_STRIDE + n] = f2bf(fmaxf(acc1[nt][r] + bias, 0.0f));
        }
    }
    asm volatile("s_wait_dscnt 0" ::: "memory");

    // ---------------- layer 2: [32x192] x [192x64]
    v8f acc2a[4], acc2b[4];
    #pragma unroll
    for (int i = 0; i < 4; ++i) { acc2a[i] = vzero; acc2b[i] = vzero; }

    #pragma unroll 1
    for (int kt = 0; kt < 6; ++kt) {
        v16bf a0, a1;
        #pragma unroll
        for (int e = 0; e < 16; ++e) {
            int kl = e + (hi8 << 1);
            a0[e] = s_h1[wave][ m       * H1_STRIDE + kt * 32 + kl];
            a1[e] = s_h1[wave][(m + 16) * H1_STRIDE + kt * 32 + kl];
        }
        #pragma unroll
        for (int nt = 0; nt < 4; ++nt) {
            v16bf b = *(const v16bf*)(w2frag + (size_t)(kt * 4 + nt) * 512 + lane * 16);
            acc2a[nt] = __builtin_amdgcn_wmma_f32_16x16x32_bf16(
                false, a0, false, b, (short)0, acc2a[nt], false, false);
            acc2b[nt] = __builtin_amdgcn_wmma_f32_16x16x32_bf16(
                false, a1, false, b, (short)0, acc2b[nt], false, false);
        }
    }

    #pragma unroll
    for (int nt = 0; nt < 4; ++nt) {
        int n = nt * 16 + m;
        float bias = b2[n];
        #pragma unroll
        for (int r = 0; r < 8; ++r) {
            s_h2[wave][(r + hi8)      * H2_STRIDE + n] = fmaxf(acc2a[nt][r] + bias, 0.0f);
            s_h2[wave][(r + hi8 + 16) * H2_STRIDE + n] = fmaxf(acc2b[nt][r] + bias, 0.0f);
        }
    }
    asm volatile("s_wait_dscnt 0" ::: "memory");

    // ---------------- layer 3 + activation: one row per lane (all 32 lanes)
    {
        float s = b3[0];
        #pragma unroll 8
        for (int n = 0; n < 64; ++n)
            s += s_h2[wave][lane * H2_STRIDE + n] * W3[n];
        float sig = 1.0f / (1.0f + expf(-s));
        int r = unit * 32 + lane;
        if (r < NROWS) scores[r] = expf(sig);
    }
}

// ------------------------------------------------------------ loss reduce
__global__ __launch_bounds__(1024)
void loss_reduce_kernel(const float* __restrict__ scores, float* __restrict__ out) {
    __shared__ float red[1024];
    float acc = 0.0f;
    for (int b = threadIdx.x; b < BPAIR; b += 1024) {
        float ps = scores[b];
        float denom = ps;
        #pragma unroll
        for (int k = 0; k < KNEG; ++k)
            denom += scores[BPAIR + b * KNEG + k];
        acc += ps / (denom + 1e-8f) + 1e-8f;
    }
    red[threadIdx.x] = acc;
    __syncthreads();
    for (int s = 512; s > 0; s >>= 1) {
        if ((int)threadIdx.x < s) red[threadIdx.x] += red[threadIdx.x + s];
        __syncthreads();
    }
    if (threadIdx.x == 0) out[0] = -red[0];
}

// ----------------------------------------------------------------- launch
extern "C" void kernel_launch(void* const* d_in, const int* in_sizes, int n_in,
                              void* d_out, int out_size, void* d_ws, size_t ws_size,
                              hipStream_t stream) {
    const float* embeds = (const float*)d_in[0];   // (L, N, H) f32
    const int*   pos    = (const int*)  d_in[1];   // (B, 2)
    const int*   neg    = (const int*)  d_in[2];   // (B, K, 2)
    const float* W1     = (const float*)d_in[3];   // (192, 576)
    const float* b1     = (const float*)d_in[4];   // (192,)
    const float* W2     = (const float*)d_in[5];   // (64, 192)
    const float* b2     = (const float*)d_in[6];   // (64,)
    const float* W3     = (const float*)d_in[7];   // (1, 64)
    const float* b3     = (const float*)d_in[8];   // (1,)
    (void)in_sizes; (void)n_in; (void)out_size; (void)ws_size;

    char* ws = (char*)d_ws;
    float*  scores = (float*)ws;                              // 170000 f32 = 680000 B
    __bf16* w1frag = (__bf16*)(ws + 680064);                  // 110592 bf16 = 221184 B
    __bf16* w2frag = (__bf16*)(ws + 680064 + 221184);         // 12288 bf16  = 24576 B
    // total workspace use: ~926 KB

    hipLaunchKernelGGL(build_w1frag, dim3(432), dim3(256), 0, stream, W1, w1frag);
    hipLaunchKernelGGL(build_w2frag, dim3(48),  dim3(256), 0, stream, W2, w2frag);

    int nblocks = (NUNITS + WAVES_PER_BLOCK - 1) / WAVES_PER_BLOCK;  // 2657
    hipLaunchKernelGGL(pair_mlp_kernel, dim3(nblocks), dim3(64), 0, stream,
                       embeds, pos, neg, b1, b2, W3, b3, w1frag, w2frag, scores);

    hipLaunchKernelGGL(loss_reduce_kernel, dim3(1), dim3(1024), 0, stream,
                       scores, (float*)d_out);
}
